// LSTMModel_42657615184500
// MI455X (gfx1250) — matrix operand order repaired and checked by reading, not compiled, not run on hardware
//
#include <hip/hip_runtime.h>
#include <hip/hip_bf16.h>

// ---------------------------------------------------------------------------
// 2-layer LSTM + FC on MI455X (gfx1250), bf16 WMMA + async-to-LDS pipeline.
//   pack/convert -> big gx GEMM (layer0) -> 128 fused recurrent steps ->
//   big gx GEMM (layer1) -> 128 steps -> FC GEMM with scattered store.
// ---------------------------------------------------------------------------

typedef unsigned short ushort_t;
typedef __attribute__((ext_vector_type(8)))  float  v8f;
typedef __attribute__((ext_vector_type(16))) __bf16 v16bf;

#define BB   256   // batch
#define SS   128   // seq len
#define HH   512   // hidden
#define INP  416   // 414 padded to 32
#define G4   2048  // 4*H
#define PITCH 40   // LDS row pitch in bf16 units (80B = 5x16B, conflict-free)

__device__ inline ushort_t f2bf(float x) {
    unsigned u = __float_as_uint(x);
    u += 0x7FFFu + ((u >> 16) & 1u);         // round-to-nearest-even
    return (ushort_t)(u >> 16);
}

__device__ inline float sigf(float x) { return 1.0f / (1.0f + __expf(-x)); }

__device__ inline v8f wmma_bf16(v16bf a, v16bf b, v8f c) {
    return __builtin_amdgcn_wmma_f32_16x16x32_bf16(false, a, false, b,
                                                   (short)0, c, false, false);
}

// Async DMA: 16B/lane global -> LDS, tracked by ASYNCcnt.
__device__ inline void async_cp16(const void* sbase, unsigned voff, unsigned loff) {
    asm volatile("global_load_async_to_lds_b128 %0, %1, %2"
                 :: "v"(loff), "v"(voff), "s"(sbase) : "memory");
}
__device__ inline void wait_async0() {
    asm volatile("s_wait_asynccnt 0" ::: "memory");
}

// A fragment 16x32 bf16 (ISA 7.12.2): lane m=lane&15, half=lane>>4,
// u[v<4]  -> K = 8*half + 2v, u[v>=4] -> K = 16 + 8*half + 2(v-4).
__device__ inline v16bf load_a_frag(const ushort_t* rowbase, int lane) {
    int m = lane & 15, half = lane >> 4;
    const ushort_t* r = rowbase + m * PITCH;
    union { v16bf v; unsigned u[8]; } f;
#pragma unroll
    for (int v = 0; v < 4; ++v) f.u[v]     = *(const unsigned*)(r + 8*half + 2*v);
#pragma unroll
    for (int v = 0; v < 4; ++v) f.u[4 + v] = *(const unsigned*)(r + 16 + 8*half + 2*v);
    return f.v;
}

// B fragment 32x16 bf16: N = lane&15, kg = lane>>4; u[v<4] -> K = 8*kg + 2v, u[v>=4] -> K = 16 + 8*kg + 2(v-4).
// LDS holds weight rows (one row per output column n), 32 K-values each.
__device__ inline v16bf load_b_frag(const ushort_t* base, int lane) {
    int n = lane & 15, kg = lane >> 4;
    const ushort_t* r = base + n * PITCH;
    union { v16bf v; unsigned u[8]; } f;
#pragma unroll
    for (int v = 0; v < 4; ++v) f.u[v]     = *(const unsigned*)(r + 8*kg + 2*v);
#pragma unroll
    for (int v = 0; v < 4; ++v) f.u[4 + v] = *(const unsigned*)(r + 16 + 8*kg + 2*v);
    return f.v;
}

// ---------------------------------------------------------------------------
// Pack / convert kernels
// ---------------------------------------------------------------------------
__global__ __launch_bounds__(256) void pack_x_kernel(const float* __restrict__ x,
                                                     ushort_t* __restrict__ xs) {
    int i = blockIdx.x * 256 + threadIdx.x;
    if (i >= SS * BB * INP) return;
    int s = i / (BB * INP);
    int rem = i - s * (BB * INP);
    int b = rem / INP, j = rem - b * INP;
    float v = 0.0f;
    if (j < 414) {
        int f = j / 207, n = j - 207 * f;
        v = x[(((size_t)b * 2 + f) * 207 + n) * SS + s];
    }
    xs[i] = f2bf(v);
}

__global__ __launch_bounds__(256) void cvt_bf16_pad_kernel(const float* __restrict__ src,
                                                           ushort_t* __restrict__ dst,
                                                           int rs, int cs, int rd, int cd) {
    int i = blockIdx.x * 256 + threadIdx.x;
    if (i >= rd * cd) return;
    int r = i / cd, c = i - r * cd;
    float v = (r < rs && c < cs) ? src[(size_t)r * cs + c] : 0.0f;
    dst[i] = f2bf(v);
}

__global__ __launch_bounds__(256) void addvec_kernel(const float* __restrict__ a,
                                                     const float* __restrict__ b,
                                                     float* __restrict__ o, int n) {
    int i = blockIdx.x * 256 + threadIdx.x;
    if (i < n) o[i] = a[i] + b[i];
}

// ---------------------------------------------------------------------------
// Tiled 128x128 GEMM:  out[M,N] = A[M,K] * W[N,K]^T (+bias)
// MODE 0: gx GEMM  -> out f32 [M,2048], acc seeded with bias[col]
// MODE 1: FC GEMM  -> acc + fc_b, scattered store into d_out [B,2,207,S]
// Block: 256 thr (8 waves). Wave = 2 Mtiles x 4 Ntiles. K staged in 32-chunks
// through double-buffered LDS via async-to-LDS DMA.
// ---------------------------------------------------------------------------
template <int MODE>
__global__ __launch_bounds__(256)
void gemm128_kernel(const ushort_t* __restrict__ A, int lda,
                    const ushort_t* __restrict__ W, int ldw,
                    const float* __restrict__ bias,
                    float* __restrict__ out, int K) {
    __shared__ ushort_t sA[2][128 * PITCH];
    __shared__ ushort_t sB[2][128 * PITCH];
    const int tid = threadIdx.x;
    const int lane = tid & 31, w = tid >> 5;
    const int half = lane >> 4, ln = lane & 15;
    const int m0 = blockIdx.x * 128, n0 = blockIdx.y * 128;
    const int mw = 32 * (w & 3), nw = 64 * (w >> 2);
    const int nk = K >> 5;

    v8f acc[2][4];
#pragma unroll
    for (int ni = 0; ni < 4; ++ni) {
        float bv = (MODE == 0) ? bias[n0 + nw + 16 * ni + ln] : 0.0f;
        v8f a;
#pragma unroll
        for (int r = 0; r < 8; ++r) a[r] = bv;
        acc[0][ni] = a; acc[1][ni] = a;
    }

    auto stage = [&](int p, int kc) {
#pragma unroll
        for (int rep = 0; rep < 2; ++rep) {          // A: 128 rows x 4 x 16B
            int task = tid + rep * 256;
            int row = task >> 2, q = task & 3;
            unsigned voff = (unsigned)((m0 + row) * lda * 2 + kc * 64 + q * 16);
            unsigned loff = (unsigned)(size_t)(&sA[p][row * PITCH]) + q * 16;
            async_cp16(A, voff, loff);
        }
#pragma unroll
        for (int rep = 0; rep < 2; ++rep) {          // B: weight rows n0..n0+127
            int task = tid + rep * 256;
            int row = task >> 2, q = task & 3;
            unsigned voff = (unsigned)((n0 + row) * ldw * 2 + kc * 64 + q * 16);
            unsigned loff = (unsigned)(size_t)(&sB[p][row * PITCH]) + q * 16;
            async_cp16(W, voff, loff);
        }
    };

    stage(0, 0);
    wait_async0();
    __syncthreads();

    for (int kc = 0; kc < nk; ++kc) {
        int p = kc & 1;
        if (kc + 1 < nk) stage(p ^ 1, kc + 1);       // DMA next chunk (overlap)
        v16bf af0 = load_a_frag(&sA[p][(mw + 0)  * PITCH], lane);
        v16bf af1 = load_a_frag(&sA[p][(mw + 16) * PITCH], lane);
#pragma unroll
        for (int ni = 0; ni < 4; ++ni) {
            v16bf bf = load_b_frag(&sB[p][(nw + 16 * ni) * PITCH], lane);
            acc[0][ni] = wmma_bf16(af0, bf, acc[0][ni]);
            acc[1][ni] = wmma_bf16(af1, bf, acc[1][ni]);
        }
        if (kc + 1 < nk) wait_async0();
        __syncthreads();
    }

#pragma unroll
    for (int mi = 0; mi < 2; ++mi)
#pragma unroll
        for (int ni = 0; ni < 4; ++ni)
#pragma unroll
            for (int r = 0; r < 8; ++r) {
                int row = m0 + mw + 16 * mi + 8 * half + r;
                int col = n0 + nw + 16 * ni + ln;
                float v = acc[mi][ni][r];
                if (MODE == 0) {
                    out[(size_t)row * G4 + col] = v;
                } else if (col < 414) {
                    v += bias[col];
                    int s = row >> 8, b = row & 255;       // row = t*256 + b
                    int c = col / 207, n = col - 207 * c;
                    out[(((size_t)b * 2 + c) * 207 + n) * SS + s] = v;
                }
            }
}

// ---------------------------------------------------------------------------
// Fused recurrent step: g = gx_t + h_{t-1} @ Whh^T, then gate math.
// Block owns 32 hidden columns for ALL 4 gates (cols {n0+g*512..+31}) over all
// 256 batch rows -> i/f/g/o for one (row,col) live in the same lane & slot of
// 4 accumulator tiles: the LSTM nonlinearity is register-local.
// Grid: 16 blocks x 256 thr; wave w owns batch rows [32w, 32w+32).
// ---------------------------------------------------------------------------
__global__ __launch_bounds__(256)
void lstm_step_kernel(const ushort_t* __restrict__ hprev,   // [256][512] bf16
                      const ushort_t* __restrict__ whh,     // [2048][512] bf16
                      const float* __restrict__ gx_t,       // [256][2048] f32
                      float* __restrict__ cbuf,             // [256][512] f32
                      ushort_t* __restrict__ hout) {        // [256][512] bf16
    __shared__ ushort_t sA[2][256 * PITCH];   // h_{t-1} rows, one K-chunk
    __shared__ ushort_t sB[2][128 * PITCH];   // 4 gates x 32 weight rows
    const int tid = threadIdx.x;
    const int lane = tid & 31, w = tid >> 5;
    const int half = lane >> 4, ln = lane & 15;
    const int n0 = 32 * blockIdx.x;
    const int mrow = 32 * w;

    v8f acc[2][4][2];   // [mi][gate][ni]
#pragma unroll
    for (int mi = 0; mi < 2; ++mi)
#pragma unroll
        for (int g = 0; g < 4; ++g)
#pragma unroll
            for (int ni = 0; ni < 2; ++ni) {
                v8f a;
#pragma unroll
                for (int r = 0; r < 8; ++r) {
                    int row = mrow + 16 * mi + 8 * half + r;
                    a[r] = gx_t[(size_t)row * G4 + g * 512 + n0 + 16 * ni + ln];
                }
                acc[mi][g][ni] = a;
            }

    auto stage = [&](int p, int kc) {
#pragma unroll
        for (int rep = 0; rep < 4; ++rep) {          // A: 256 rows x 4 x 16B
            int task = tid + rep * 256;
            int row = task >> 2, q = task & 3;
            unsigned voff = (unsigned)(row * HH * 2 + kc * 64 + q * 16);
            unsigned loff = (unsigned)(size_t)(&sA[p][row * PITCH]) + q * 16;
            async_cp16(hprev, voff, loff);
        }
#pragma unroll
        for (int rep = 0; rep < 2; ++rep) {          // B: 128 weight rows
            int task = tid + rep * 256;
            int rowi = task >> 2, q = task & 3;
            int gate = rowi >> 5, j = rowi & 31;
            int n = gate * 512 + n0 + j;
            unsigned voff = (unsigned)(n * HH * 2 + kc * 64 + q * 16);
            unsigned loff = (unsigned)(size_t)(&sB[p][rowi * PITCH]) + q * 16;
            async_cp16(whh, voff, loff);
        }
    };

    stage(0, 0);
    wait_async0();
    __syncthreads();

    for (int kc = 0; kc < 16; ++kc) {
        int p = kc & 1;
        if (kc + 1 < 16) stage(p ^ 1, kc + 1);
        v16bf af0 = load_a_frag(&sA[p][(mrow + 0)  * PITCH], lane);
        v16bf af1 = load_a_frag(&sA[p][(mrow + 16) * PITCH], lane);
#pragma unroll
        for (int g = 0; g < 4; ++g) {
            v16bf b0 = load_b_frag(&sB[p][(g * 32 + 0)  * PITCH], lane);
            v16bf b1 = load_b_frag(&sB[p][(g * 32 + 16) * PITCH], lane);
            acc[0][g][0] = wmma_bf16(af0, b0, acc[0][g][0]);
            acc[1][g][0] = wmma_bf16(af1, b0, acc[1][g][0]);
            acc[0][g][1] = wmma_bf16(af0, b1, acc[0][g][1]);
            acc[1][g][1] = wmma_bf16(af1, b1, acc[1][g][1]);
        }
        if (kc + 1 < 16) wait_async0();
        __syncthreads();
    }

    // Gate math: fully register-local across the four gates.
#pragma unroll
    for (int mi = 0; mi < 2; ++mi)
#pragma unroll
        for (int ni = 0; ni < 2; ++ni)
#pragma unroll
            for (int r = 0; r < 8; ++r) {
                int row = mrow + 16 * mi + 8 * half + r;
                int hcol = n0 + 16 * ni + ln;
                float gi = acc[mi][0][ni][r];
                float gf = acc[mi][1][ni][r];
                float gg = acc[mi][2][ni][r];
                float go = acc[mi][3][ni][r];
                float cp = cbuf[(size_t)row * HH + hcol];
                float cn = sigf(gf) * cp + sigf(gi) * tanhf(gg);
                float hv = sigf(go) * tanhf(cn);
                cbuf[(size_t)row * HH + hcol] = cn;
                hout[(size_t)row * HH + hcol] = f2bf(hv);
            }
}

// ---------------------------------------------------------------------------
extern "C" void kernel_launch(void* const* d_in, const int* in_sizes, int n_in,
                              void* d_out, int out_size, void* d_ws, size_t ws_size,
                              hipStream_t stream) {
    (void)in_sizes; (void)n_in; (void)out_size; (void)ws_size;
    const float* x     = (const float*)d_in[0];
    const float* w_ih0 = (const float*)d_in[1];
    const float* w_hh0 = (const float*)d_in[2];
    const float* b_ih0 = (const float*)d_in[3];
    const float* b_hh0 = (const float*)d_in[4];
    const float* w_ih1 = (const float*)d_in[5];
    const float* w_hh1 = (const float*)d_in[6];
    const float* b_ih1 = (const float*)d_in[7];
    const float* b_hh1 = (const float*)d_in[8];
    const float* fc_w  = (const float*)d_in[9];
    const float* fc_b  = (const float*)d_in[10];

    char* ws = (char*)d_ws;
    size_t o = 0;
    auto take = [&](size_t bytes) { size_t p = o; o += (bytes + 255) & ~(size_t)255; return p; };
    ushort_t* xs    = (ushort_t*)(ws + take((size_t)SS * BB * INP * 2));
    ushort_t* h1seq = (ushort_t*)(ws + take((size_t)SS * BB * HH * 2));
    ushort_t* h2seq = (ushort_t*)(ws + take((size_t)SS * BB * HH * 2));
    ushort_t* hzero = (ushort_t*)(ws + take((size_t)BB * HH * 2));
    ushort_t* wih0b = (ushort_t*)(ws + take((size_t)G4 * INP * 2));
    ushort_t* whh0b = (ushort_t*)(ws + take((size_t)G4 * HH * 2));
    ushort_t* wih1b = (ushort_t*)(ws + take((size_t)G4 * HH * 2));
    ushort_t* whh1b = (ushort_t*)(ws + take((size_t)G4 * HH * 2));
    ushort_t* fcwb  = (ushort_t*)(ws + take((size_t)512 * HH * 2));
    float*    bias0 = (float*)(ws + take((size_t)G4 * 4));
    float*    bias1 = (float*)(ws + take((size_t)G4 * 4));
    float*    cbuf  = (float*)(ws + take((size_t)BB * HH * 4));
    float*    gx    = (float*)(ws + take((size_t)SS * BB * G4 * 4));

    // --- pack & convert ---
    pack_x_kernel<<<(SS * BB * INP + 255) / 256, 256, 0, stream>>>(x, xs);
    cvt_bf16_pad_kernel<<<(G4 * INP + 255) / 256, 256, 0, stream>>>(w_ih0, wih0b, G4, 414, G4, INP);
    cvt_bf16_pad_kernel<<<(G4 * HH + 255) / 256, 256, 0, stream>>>(w_hh0, whh0b, G4, HH, G4, HH);
    cvt_bf16_pad_kernel<<<(G4 * HH + 255) / 256, 256, 0, stream>>>(w_ih1, wih1b, G4, HH, G4, HH);
    cvt_bf16_pad_kernel<<<(G4 * HH + 255) / 256, 256, 0, stream>>>(w_hh1, whh1b, G4, HH, G4, HH);
    cvt_bf16_pad_kernel<<<(512 * HH + 255) / 256, 256, 0, stream>>>(fc_w, fcwb, 414, HH, 512, HH);
    addvec_kernel<<<(G4 + 255) / 256, 256, 0, stream>>>(b_ih0, b_hh0, bias0, G4);
    addvec_kernel<<<(G4 + 255) / 256, 256, 0, stream>>>(b_ih1, b_hh1, bias1, G4);
    hipMemsetAsync(hzero, 0, (size_t)BB * HH * 2, stream);
    hipMemsetAsync(cbuf, 0, (size_t)BB * HH * 4, stream);

    // --- layer 0 ---
    gemm128_kernel<0><<<dim3(SS * BB / 128, G4 / 128), 256, 0, stream>>>(
        xs, INP, wih0b, INP, bias0, gx, INP);
    for (int t = 0; t < SS; ++t) {
        const ushort_t* hp = (t == 0) ? hzero : h1seq + (size_t)(t - 1) * BB * HH;
        lstm_step_kernel<<<16, 256, 0, stream>>>(
            hp, whh0b, gx + (size_t)t * BB * G4, cbuf, h1seq + (size_t)t * BB * HH);
    }

    // --- layer 1 ---
    hipMemsetAsync(cbuf, 0, (size_t)BB * HH * 4, stream);
    gemm128_kernel<0><<<dim3(SS * BB / 128, G4 / 128), 256, 0, stream>>>(
        h1seq, HH, wih1b, HH, bias1, gx, HH);
    for (int t = 0; t < SS; ++t) {
        const ushort_t* hp = (t == 0) ? hzero : h2seq + (size_t)(t - 1) * BB * HH;
        lstm_step_kernel<<<16, 256, 0, stream>>>(
            hp, whh1b, gx + (size_t)t * BB * G4, cbuf, h2seq + (size_t)t * BB * HH);
    }

    // --- FC head with fused transpose-scatter ---
    gemm128_kernel<1><<<dim3(SS * BB / 128, 512 / 128), 256, 0, stream>>>(
        h2seq, HH, fcwb, HH, fc_b, (float*)d_out, HH);
}
